// SNPCFGFixedCostRewardA2C_20048907338320
// MI455X (gfx1250) — matrix operand order repaired and checked by reading, not compiled, not run on hardware
//
#include <hip/hip_runtime.h>
#include <math.h>

// Problem constants (from reference): NT=60, T=120, S=512, V=10000, B=128, N=40
#define NTN 60
#define TT  120
#define SS  512
#define VV  10000
#define BB  128
#define NN  40
#define ROW_STRIDE  64           // NT padded to 64 floats (256B rows)
#define CELL_STRIDE (BB * ROW_STRIDE)
#define NEGBIG (-1e30f)

typedef __attribute__((ext_vector_type(16))) __bf16 bf16x16;
typedef __attribute__((ext_vector_type(8)))  float  f32x8;

// Triangular chart: level w (1..N) has N-w+1 cells; cells of level w start at off(w).
__device__ __forceinline__ long cell_off(int w) {
    return (long)(w - 1) * (NN + 1) - (long)(w - 1) * w / 2;
}

// Split-bf16 ("bf16x2") operand decomposition: v ~= hi + lo with ~16 mantissa
// bits total; hi*hi + hi*lo + lo*hi in f32 accumulators gives ~f32 accuracy.
__device__ __forceinline__ void pack8s(bf16x16& hi, bf16x16& lo, int base,
                                       float4 x, float4 y) {
    float v; __bf16 h;
    v = x.x; h = (__bf16)v; hi[base + 0] = h; lo[base + 0] = (__bf16)(v - (float)h);
    v = x.y; h = (__bf16)v; hi[base + 1] = h; lo[base + 1] = (__bf16)(v - (float)h);
    v = x.z; h = (__bf16)v; hi[base + 2] = h; lo[base + 2] = (__bf16)(v - (float)h);
    v = x.w; h = (__bf16)v; hi[base + 3] = h; lo[base + 3] = (__bf16)(v - (float)h);
    v = y.x; h = (__bf16)v; hi[base + 4] = h; lo[base + 4] = (__bf16)(v - (float)h);
    v = y.y; h = (__bf16)v; hi[base + 5] = h; lo[base + 5] = (__bf16)(v - (float)h);
    v = y.z; h = (__bf16)v; hi[base + 6] = h; lo[base + 6] = (__bf16)(v - (float)h);
    v = y.w; h = (__bf16)v; hi[base + 7] = h; lo[base + 7] = (__bf16)(v - (float)h);
}

__device__ __forceinline__ void pack8s_exp(bf16x16& hi, bf16x16& lo, int base,
                                           float4 x, float4 y, float rm) {
    float v; __bf16 h;
    v = __expf(x.x - rm); h = (__bf16)v; hi[base + 0] = h; lo[base + 0] = (__bf16)(v - (float)h);
    v = __expf(x.y - rm); h = (__bf16)v; hi[base + 1] = h; lo[base + 1] = (__bf16)(v - (float)h);
    v = __expf(x.z - rm); h = (__bf16)v; hi[base + 2] = h; lo[base + 2] = (__bf16)(v - (float)h);
    v = __expf(x.w - rm); h = (__bf16)v; hi[base + 3] = h; lo[base + 3] = (__bf16)(v - (float)h);
    v = __expf(y.x - rm); h = (__bf16)v; hi[base + 4] = h; lo[base + 4] = (__bf16)(v - (float)h);
    v = __expf(y.y - rm); h = (__bf16)v; hi[base + 5] = h; lo[base + 5] = (__bf16)(v - (float)h);
    v = __expf(y.z - rm); h = (__bf16)v; hi[base + 6] = h; lo[base + 6] = (__bf16)(v - (float)h);
    v = __expf(y.w - rm); h = (__bf16)v; hi[base + 7] = h; lo[base + 7] = (__bf16)(v - (float)h);
}

#define WMMA_BF16(A, B, C) \
    __builtin_amdgcn_wmma_f32_16x16x32_bf16(false, A, false, B, (short)0, C, false, false)

// ---------------------------------------------------------------------------
// GEMM: C = op(A[M,K] @ W[N,K]^T + bias).  K multiple of 32, lda/ldw mult of 4.
// flags: 2 = relu, 4 = add residual res[M,N].  One wave per 16x16 tile.
// Split-bf16 operands (3 WMMAs / K-step) -> ~f32 accuracy, f32 accumulate.
// Edge rows/cols are handled by clamped (in-bounds) loads + guarded stores.
// ---------------------------------------------------------------------------
__global__ void gemm_bf16_wmma(const float* __restrict__ A, int lda,
                               const float* __restrict__ W, int ldw,
                               const float* __restrict__ bias,
                               const float* res, int ldres,
                               float* C, int ldc,
                               int M, int N, int K, int flags)
{
    const int lane = threadIdx.x & 31;
    const int mrow = lane & 15;
    const int half = lane >> 4;
    const int m0 = blockIdx.y << 4;
    const int n0 = blockIdx.x << 4;
    const int arow = min(m0 + mrow, M - 1);   // clamped: garbage rows are never stored
    const int bcol = min(n0 + mrow, N - 1);
    f32x8 acc = {0.f, 0.f, 0.f, 0.f, 0.f, 0.f, 0.f, 0.f};

    const float* Arow = A + (size_t)arow * lda;
    const float* Wrow = W + (size_t)bcol * ldw;
    for (int kk = 0; kk < K; kk += 32) {
        const float4* ap = (const float4*)(Arow + kk);
        const float4 a0 = ap[half * 4 + 0];
        const float4 a1 = ap[half * 4 + 1];
        const float4 a2 = ap[half * 4 + 2];
        const float4 a3 = ap[half * 4 + 3];
        const float4* bp = (const float4*)(Wrow + kk + half * 16);
        const float4 b0 = bp[0], b1 = bp[1], b2 = bp[2], b3 = bp[3];
        bf16x16 ah, al, bh, bl;
        pack8s(ah, al, 0, a0, a1); pack8s(ah, al, 8, a2, a3);
        pack8s(bh, bl, 0, b0, b1); pack8s(bh, bl, 8, b2, b3);
        acc = WMMA_BF16(ah, bh, acc);
        acc = WMMA_BF16(ah, bl, acc);
        acc = WMMA_BF16(al, bh, acc);
    }
#pragma unroll
    for (int v = 0; v < 8; ++v) {
        const int m = m0 + half * 8 + v;
        const int n = n0 + mrow;
        if (m < M && n < N) {
            float o = acc[v];
            if (bias) o += bias[n];
            if (flags & 2) o = fmaxf(o, 0.f);
            if (flags & 4) o += res[(size_t)m * ldres + n];
            C[(size_t)m * ldc + n] = o;
        }
    }
}

// ---------------------------------------------------------------------------
// Dual log-matmul: out = log(exp(X - rowmax) @ P) + rowmax for PL and PR,
// stored into chart cells LC/RC at level w.  X padded with -1e30 (exp -> 0),
// PLT/PRT are transposed, zero-padded probability tables [n][k] (ld = ldk).
// rows is a multiple of 16; K in {64, 128}.  Block = 128 threads = 4 N-tiles.
// Split-bf16 on both operands: 6 WMMAs per K-step for the two outputs.
// rowmode 0: row r = b*N + i (leaf);  rowmode 1: row r = i*B + b.
// ---------------------------------------------------------------------------
__global__ void logmm_dual_wmma(const float* __restrict__ X, int ldx,
                                const float* __restrict__ rmax,
                                const float* __restrict__ PLT,
                                const float* __restrict__ PRT, int ldk,
                                int K, int rows,
                                float* __restrict__ LC, float* __restrict__ RC,
                                int w, int rowmode)
{
    const int lane = threadIdx.x & 31;
    const int mrow = lane & 15;
    const int half = lane >> 4;
    const int m0 = blockIdx.y << 4;
    const int n0 = (threadIdx.x >> 5) << 4;   // wave id selects N-tile (0..3)
    const int arow = m0 + mrow;               // rows multiple of 16 -> in range
    const int bcol = n0 + mrow;               // <= 63, PT tables have 64 rows
    const float rm = rmax[arow];
    f32x8 accL = {0.f, 0.f, 0.f, 0.f, 0.f, 0.f, 0.f, 0.f};
    f32x8 accR = {0.f, 0.f, 0.f, 0.f, 0.f, 0.f, 0.f, 0.f};

    const float* Xrow = X + (size_t)arow * ldx;
    const float* Lrow = PLT + (size_t)bcol * ldk;
    const float* Rrow = PRT + (size_t)bcol * ldk;
    for (int kk = 0; kk < K; kk += 32) {
        const float4* ap = (const float4*)(Xrow + kk);
        const float4 a0 = ap[half * 4 + 0];
        const float4 a1 = ap[half * 4 + 1];
        const float4 a2 = ap[half * 4 + 2];
        const float4 a3 = ap[half * 4 + 3];
        const float4* lp = (const float4*)(Lrow + kk + half * 16);
        const float4 l0 = lp[0], l1 = lp[1], l2 = lp[2], l3 = lp[3];
        const float4* rp = (const float4*)(Rrow + kk + half * 16);
        const float4 r0 = rp[0], r1 = rp[1], r2 = rp[2], r3 = rp[3];
        bf16x16 ah, al, lh, ll, rh, rl;
        pack8s_exp(ah, al, 0, a0, a1, rm); pack8s_exp(ah, al, 8, a2, a3, rm);
        pack8s(lh, ll, 0, l0, l1); pack8s(lh, ll, 8, l2, l3);
        pack8s(rh, rl, 0, r0, r1); pack8s(rh, rl, 8, r2, r3);
        accL = WMMA_BF16(ah, lh, accL);
        accL = WMMA_BF16(ah, ll, accL);
        accL = WMMA_BF16(al, lh, accL);
        accR = WMMA_BF16(ah, rh, accR);
        accR = WMMA_BF16(ah, rl, accR);
        accR = WMMA_BF16(al, rh, accR);
    }
    const long off = cell_off(w);
#pragma unroll
    for (int v = 0; v < 8; ++v) {
        const int r = m0 + half * 8 + v;
        const int n = n0 + mrow;
        if (n < NTN) {
            int b, i;
            if (rowmode == 0) { b = r / NN; i = r % NN; }
            else              { i = r / BB; b = r % BB; }
            const long cell = off + i;
            const float mx = rmax[r];
            LC[cell * CELL_STRIDE + b * ROW_STRIDE + n] = __logf(accL[v]) + mx;
            RC[cell * CELL_STRIDE + b * ROW_STRIDE + n] = __logf(accR[v]) + mx;
        }
    }
}

// ---------------------------------------------------------------------------
// beta[i,b,nt] = logsumexp_k( Lc[k,i][b,nt] + Rc[w-k,i+k][b,nt] ), k=1..w-1
// One block per (i,b) row; pads nt=60..63 with -1e30; fused row-max -> brmax.
// Distance-2 prefetch (global_prefetch_b8) hides L2 latency across splits.
// ---------------------------------------------------------------------------
__global__ void beta_kernel(const float* __restrict__ LC, const float* __restrict__ RC,
                            float* __restrict__ beta, float* __restrict__ brmax, int w)
{
    const int row = blockIdx.x;          // row = i*BB + b
    const int i = row / BB;
    const int b = row % BB;
    const int nt = threadIdx.x;          // 64 threads, nt<60 active
    float val = NEGBIG;
    if (nt < NTN) {
        float mx = NEGBIG, s = 0.f;
        for (int k = 1; k < w; ++k) {
            if (k + 2 < w) {
                __builtin_prefetch(LC + (cell_off(k + 2) + i) * CELL_STRIDE
                                      + b * ROW_STRIDE + nt, 0, 1);
                __builtin_prefetch(RC + (cell_off(w - k - 2) + i + k + 2) * CELL_STRIDE
                                      + b * ROW_STRIDE + nt, 0, 1);
            }
            const float lv = LC[(cell_off(k) + i) * CELL_STRIDE + b * ROW_STRIDE + nt];
            const float rv = RC[(cell_off(w - k) + i + k) * CELL_STRIDE + b * ROW_STRIDE + nt];
            const float x = lv + rv;
            if (x > mx) { s = s * __expf(mx - x) + 1.f; mx = x; }
            else        { s += __expf(x - mx); }
        }
        val = mx + __logf(s);
    }
    beta[(size_t)row * ROW_STRIDE + nt] = val;   // pad lanes store -1e30 (exp -> 0)
    __shared__ float red[64];
    red[nt] = val;
    __syncthreads();
    for (int st = 32; st > 0; st >>= 1) {
        if (nt < st) red[nt] = fmaxf(red[nt], red[nt + st]);
        __syncthreads();
    }
    if (nt == 0) brmax[row] = red[0];
}

// Per-row max, one thread per row.
__global__ void rowmax_kernel(const float* __restrict__ X, float* __restrict__ out,
                              int rows, int K, int ld)
{
    const int r = blockIdx.x * blockDim.x + threadIdx.x;
    if (r >= rows) return;
    float m = NEGBIG;
    for (int k = 0; k < K; ++k) m = fmaxf(m, X[(size_t)r * ld + k]);
    out[r] = m;
}

// Per-row logsumexp (vocab axis), one block per row.
__global__ void row_lse_kernel(const float* __restrict__ X, float* __restrict__ out,
                               int K, int ld)
{
    const int row = blockIdx.x;
    const int t = threadIdx.x;           // 256
    __shared__ float red[256];
    float mx = NEGBIG;
    for (int k = t; k < K; k += 256) mx = fmaxf(mx, X[(size_t)row * ld + k]);
    red[t] = mx; __syncthreads();
    for (int st = 128; st > 0; st >>= 1) {
        if (t < st) red[t] = fmaxf(red[t], red[t + st]);
        __syncthreads();
    }
    mx = red[0]; __syncthreads();
    float s = 0.f;
    for (int k = t; k < K; k += 256) s += __expf(X[(size_t)row * ld + k] - mx);
    red[t] = s; __syncthreads();
    for (int st = 128; st > 0; st >>= 1) {
        if (t < st) red[t] += red[t + st];
        __syncthreads();
    }
    if (t == 0) out[row] = mx + __logf(red[0]);
}

__global__ void add_kernel(const float* __restrict__ a, const float* __restrict__ b,
                           float* __restrict__ c, int n)
{
    const int i = blockIdx.x * blockDim.x + threadIdx.x;
    if (i < n) c[i] = a[i] + b[i];
}

// vocabT[v][k] = vocab[k][v]  (coalesced writes)
__global__ void transpose_kernel(const float* __restrict__ src, float* __restrict__ dst)
{
    const int idx = blockIdx.x * blockDim.x + threadIdx.x;
    if (idx >= SS * VV) return;
    const int v = idx / SS;
    const int k = idx % SS;
    dst[idx] = src[(size_t)k * VV + v];
}

__global__ void zero_kernel(float* __restrict__ p, int n)
{
    const int i = blockIdx.x * blockDim.x + threadIdx.x;
    if (i < n) p[i] = 0.f;
}

// unary[b,n,t] = term_logits[t, word[b,n]] - lse[t]; stride 128, pad = -1e30
__global__ void unary_kernel(const float* __restrict__ logits, const float* __restrict__ lse,
                             const int* __restrict__ word, float* __restrict__ unary)
{
    const int idx = blockIdx.x * blockDim.x + threadIdx.x;
    if (idx >= BB * NN * 128) return;
    const int t = idx & 127;
    const int n = (idx >> 7) % NN;
    const int b = idx / (128 * NN);
    float v = NEGBIG;
    if (t < TT) v = logits[(size_t)t * VV + word[b * NN + n]] - lse[t];
    unary[idx] = v;
}

// Column softmax over axis 0 (180 rows) of S[180x60]; writes transposed padded
// tables: PmT[n][k] (ld 64, k<60 real) and PpT[n][k] (ld 128, k<120 real).
__global__ void softmax_pack_kernel(const float* __restrict__ S,
                                    float* __restrict__ PmT, float* __restrict__ PpT)
{
    const int col = blockIdx.x;          // 0..59  (n index)
    const int t = threadIdx.x;           // 64
    __shared__ float red[64];
    float mx = NEGBIG;
    for (int r = t; r < NTN + TT; r += 64) mx = fmaxf(mx, S[r * NTN + col]);
    red[t] = mx; __syncthreads();
    for (int st = 32; st > 0; st >>= 1) {
        if (t < st) red[t] = fmaxf(red[t], red[t + st]);
        __syncthreads();
    }
    mx = red[0]; __syncthreads();
    float s = 0.f;
    for (int r = t; r < NTN + TT; r += 64) s += __expf(S[r * NTN + col] - mx);
    red[t] = s; __syncthreads();
    for (int st = 32; st > 0; st >>= 1) {
        if (t < st) red[t] += red[t + st];
        __syncthreads();
    }
    const float inv = 1.f / red[0];
    __syncthreads();
    for (int r = t; r < NTN + TT; r += 64) {
        const float p = __expf(S[r * NTN + col] - mx) * inv;
        if (r < NTN) PmT[col * 64 + r] = p;
        else         PpT[col * 128 + (r - NTN)] = p;
    }
}

// In-place log_softmax of a short vector (single block of 64).
__global__ void log_softmax_vec_kernel(float* __restrict__ x, int n)
{
    const int t = threadIdx.x;           // 64
    __shared__ float red[64];
    float v = (t < n) ? x[t] : NEGBIG;
    red[t] = v; __syncthreads();
    for (int st = 32; st > 0; st >>= 1) {
        if (t < st) red[t] = fmaxf(red[t], red[t + st]);
        __syncthreads();
    }
    const float mx = red[0]; __syncthreads();
    float e = (t < n) ? __expf(v - mx) : 0.f;
    red[t] = e; __syncthreads();
    for (int st = 32; st > 0; st >>= 1) {
        if (t < st) red[t] += red[t + st];
        __syncthreads();
    }
    const float lse = mx + __logf(red[0]);
    if (t < n) x[t] = v - lse;
}

// partition[b] = logsumexp_n( root[n] + betaTop[b,n] )
__global__ void partition_kernel(const float* __restrict__ rootv,
                                 const float* __restrict__ betaTop,
                                 float* __restrict__ out)
{
    const int b = blockIdx.x;
    const int t = threadIdx.x;           // 64
    __shared__ float red[64];
    float v = (t < NTN) ? rootv[t] + betaTop[b * ROW_STRIDE + t] : NEGBIG;
    red[t] = v; __syncthreads();
    for (int st = 32; st > 0; st >>= 1) {
        if (t < st) red[t] = fmaxf(red[t], red[t + st]);
        __syncthreads();
    }
    const float mx = red[0]; __syncthreads();
    float e = (t < NTN) ? __expf(v - mx) : 0.f;
    red[t] = e; __syncthreads();
    for (int st = 32; st > 0; st >>= 1) {
        if (t < st) red[t] += red[t + st];
        __syncthreads();
    }
    if (t == 0) out[b] = mx + __logf(red[0]);
}

__global__ void loss_kernel(float* __restrict__ out)
{
    const int t = threadIdx.x;           // 128
    __shared__ float red[128];
    red[t] = out[t]; __syncthreads();
    for (int st = 64; st > 0; st >>= 1) {
        if (t < st) red[t] += red[t + st];
        __syncthreads();
    }
    if (t == 0) out[BB] = -red[0] / (float)BB;
}

// ---------------------------------------------------------------------------
// Probe (confirmed lowering in round 2): pure-f32 WMMA V_WMMA_F32_16X16X4_F32.
// Kept as documentation of the available f32 path.
// ---------------------------------------------------------------------------
#if defined(__gfx1250__) && __has_builtin(__builtin_amdgcn_wmma_f32_16x16x4_f32)
typedef __attribute__((ext_vector_type(2))) float f32x2;
__global__ void probe_wmma_f32x4(const float* __restrict__ A,
                                 const float* __restrict__ B,
                                 float* __restrict__ C)
{
    const int lane = threadIdx.x & 31;
    f32x2 a, b;
    a.x = A[lane]; a.y = A[lane + 32];
    b.x = B[lane]; b.y = B[lane + 32];
    f32x8 c = {0.f, 0.f, 0.f, 0.f, 0.f, 0.f, 0.f, 0.f};
    c = __builtin_amdgcn_wmma_f32_16x16x4_f32(false, a, false, b, (short)0, c, false, false);
    *(f32x8*)(C + lane * 8) = c;
}
#endif

// ---------------------------------------------------------------------------
extern "C" void kernel_launch(void* const* d_in, const int* in_sizes, int n_in,
                              void* d_out, int out_size, void* d_ws, size_t ws_size,
                              hipStream_t stream)
{
    (void)in_sizes; (void)n_in; (void)out_size; (void)ws_size;
    const int*   word     = (const int*)d_in[0];
    const float* root_emb = (const float*)d_in[1];
    const float* rse      = (const float*)d_in[2];
    const float* vocab    = (const float*)d_in[3];
    const float* W0       = (const float*)d_in[4];
    const float* b0       = (const float*)d_in[5];
    const float* W1       = (const float*)d_in[6];
    const float* b1       = (const float*)d_in[7];
    const float* W2       = (const float*)d_in[8];
    const float* b2       = (const float*)d_in[9];
    const float* lW       = (const float*)d_in[10];
    const float* lb       = (const float*)d_in[11];
    const float* rW       = (const float*)d_in[12];
    const float* rb       = (const float*)d_in[13];
    const float* pW       = (const float*)d_in[14];
    const float* pb       = (const float*)d_in[15];
    float* out = (float*)d_out;

    // Workspace carve-up (floats); total ~85 MB (chart + vocabT dominate)
    float* P = (float*)d_ws;
    size_t o = 0;
    auto alloc = [&](size_t n) { float* r = P + o; o += (n + 63) & ~63ull; return r; };
    float* hA     = alloc((size_t)TT * SS);
    float* hB     = alloc((size_t)TT * SS);
    float* t1     = alloc((size_t)TT * SS);
    float* hpe    = alloc((size_t)TT * SS);
    float* vocabT = alloc((size_t)VV * SS);
    float* tlog   = alloc((size_t)TT * VV);
    float* tlse   = alloc(128);
    float* unry   = alloc((size_t)BB * NN * 128);
    float* urmax  = alloc((size_t)BB * NN);
    float* lact   = alloc((size_t)(NTN + TT) * SS);
    float* ract   = alloc((size_t)(NTN + TT) * SS);
    float* par1   = alloc((size_t)NTN * SS);
    float* ls     = alloc((size_t)(NTN + TT) * NTN);
    float* rs     = alloc((size_t)(NTN + TT) * NTN);
    float* LmT    = alloc(64 * 64);      // contiguous block: LmT,LpT,RmT,RpT
    float* LpT    = alloc(64 * 128);
    float* RmT    = alloc(64 * 64);
    float* RpT    = alloc(64 * 128);
    float* rootv  = alloc(64);
    float* beta   = alloc((size_t)NN * BB * ROW_STRIDE);
    float* brmax  = alloc((size_t)NN * BB);
    float* LC     = alloc(820ull * CELL_STRIDE);
    float* RC     = alloc(820ull * CELL_STRIDE);

    auto gemm = [&](const float* A, int lda, const float* W, int ldw,
                    const float* bias, const float* res, int ldres,
                    float* C, int ldc, int M, int N, int K, int flags) {
        dim3 g((N + 15) / 16, (M + 15) / 16);
        gemm_bf16_wmma<<<g, 32, 0, stream>>>(A, lda, W, ldw, bias, res, ldres,
                                             C, ldc, M, N, K, flags);
    };

    // ---- Grammar head ----
    transpose_kernel<<<(SS * VV + 255) / 256, 256, 0, stream>>>(vocab, vocabT);
    const float* term_emb = rse + (size_t)NTN * SS;
    gemm(term_emb, SS, W0, SS, b0, nullptr, 0, hA, SS, TT, SS, SS, 0);
    float* hc = hA; float* hn = hB;
    for (int i = 0; i < 3; ++i) {
        gemm(hc, SS, W1 + (size_t)i * SS * SS, SS, b1 + i * SS, nullptr, 0,
             t1, SS, TT, SS, SS, 2);
        gemm(t1, SS, W2 + (size_t)i * SS * SS, SS, b2 + i * SS, hc, SS,
             hn, SS, TT, SS, SS, 2 | 4);
        float* tmp = hc; hc = hn; hn = tmp;
    }
    add_kernel<<<(TT * SS + 255) / 256, 256, 0, stream>>>(hc, term_emb, hpe, TT * SS);
    gemm(hpe, SS, vocabT, SS, nullptr, nullptr, 0, tlog, VV, TT, VV, SS, 0);
    row_lse_kernel<<<TT, 256, 0, stream>>>(tlog, tlse, VV, VV);
    unary_kernel<<<(BB * NN * 128 + 255) / 256, 256, 0, stream>>>(tlog, tlse, word, unry);
    rowmax_kernel<<<(BB * NN + 255) / 256, 256, 0, stream>>>(unry, urmax, BB * NN, 128, 128);

    // left/right/parent branches
    gemm(rse, SS, lW, SS, lb, rse, SS, lact, SS, NTN + TT, SS, SS, 2 | 4);
    gemm(rse, SS, rW, SS, rb, rse, SS, ract, SS, NTN + TT, SS, SS, 2 | 4);
    gemm(rse, SS, pW, SS, pb, rse, SS, par1, SS, NTN, SS, SS, 2 | 4);
    gemm(lact, SS, par1, SS, nullptr, nullptr, 0, ls, NTN, NTN + TT, NTN, SS, 0);
    gemm(ract, SS, par1, SS, nullptr, nullptr, 0, rs, NTN, NTN + TT, NTN, SS, 0);
    gemm(root_emb, SS, rse, SS, nullptr, nullptr, 0, rootv, NTN, 1, NTN, SS, 0);
    // zero the 4 packed tables (contiguous block starting at LmT), then pack
    zero_kernel<<<(2 * (64 * 64 + 64 * 128) + 255) / 256, 256, 0, stream>>>(
        LmT, 2 * (64 * 64 + 64 * 128));
    softmax_pack_kernel<<<NTN, 64, 0, stream>>>(ls, LmT, LpT);
    softmax_pack_kernel<<<NTN, 64, 0, stream>>>(rs, RmT, RpT);
    log_softmax_vec_kernel<<<1, 64, 0, stream>>>(rootv, NTN);

    // ---- Inside (CKY) ----
    // Level 1: Lc[1]/Rc[1] = logmm(unary, left_p/right_p), rows = (b, i)
    {
        dim3 g(1, (BB * NN) / 16);
        logmm_dual_wmma<<<g, 128, 0, stream>>>(unry, 128, urmax, LpT, RpT, 128,
                                               128, BB * NN, LC, RC, 1, 0);
    }
    for (int w = 2; w <= NN; ++w) {
        const int m = NN - w + 1;
        beta_kernel<<<m * BB, 64, 0, stream>>>(LC, RC, beta, brmax, w);
        if (w < NN) {
            dim3 g(1, (m * BB) / 16);
            logmm_dual_wmma<<<g, 128, 0, stream>>>(beta, ROW_STRIDE, brmax, LmT, RmT, 64,
                                                   64, m * BB, LC, RC, w, 1);
        }
    }
    partition_kernel<<<BB, 64, 0, stream>>>(rootv, beta, out);
    loss_kernel<<<1, BB, 0, stream>>>(out);
}
